// Attention_31207232372930
// MI455X (gfx1250) — matrix operand order repaired and checked by reading, not compiled, not run on hardware
//
#include <hip/hip_runtime.h>
#include <hip/hip_bf16.h>
#include <stdint.h>

// ---------------- problem constants ----------------
#define D_MODEL   1024
#define NUM_HEADS 16
#define DK        64
#define SEQ       2048
#define BATCH     4
#define ROWS      (BATCH * SEQ)          // 8192 token rows

// gfx1250 async global->LDS path (ASYNCcnt) if the toolchain exposes it
#if defined(__has_builtin)
#  if __has_builtin(__builtin_amdgcn_global_load_async_to_lds_b128)
#    define HAVE_ASYNC_LDS 1
#  endif
#endif
#ifndef HAVE_ASYNC_LDS
#  define HAVE_ASYNC_LDS 0
#endif

typedef __attribute__((ext_vector_type(16))) __bf16 v16bf;
typedef __attribute__((ext_vector_type(8)))  float  v8f;

#if HAVE_ASYNC_LDS
typedef int v4i __attribute__((vector_size(16)));
typedef __attribute__((address_space(1))) v4i* gptr_v4i;   // global
typedef __attribute__((address_space(3))) v4i* lptr_v4i;   // LDS

__device__ __forceinline__ void async_cp16(const void* g, void* l) {
    __builtin_amdgcn_global_load_async_to_lds_b128((gptr_v4i)(g), (lptr_v4i)(l), 0, 0);
}
#endif

union Frag {
    v16bf          v;       // 16 bf16 = 32B = 8 VGPRs (WMMA A/B operand)
    uint4          q[2];
    unsigned short us[16];
};

__device__ __forceinline__ unsigned short f2bfu(float f) {
    unsigned int u = __builtin_bit_cast(unsigned int, f);
    u += 0x7FFFu + ((u >> 16) & 1u);
    return (unsigned short)(u >> 16);
}

__device__ __forceinline__ v8f zero8() {
    v8f z = {0.f, 0.f, 0.f, 0.f, 0.f, 0.f, 0.f, 0.f};
    return z;
}

__device__ __forceinline__ v8f wmma_bf16(const Frag& a, const Frag& b, v8f c) {
    return __builtin_amdgcn_wmma_f32_16x16x32_bf16(
        false, a.v, false, b.v, (short)0, c, false, false);
}

// ---------------- 1) elementwise f32 -> bf16 ----------------
__global__ __launch_bounds__(256) void k_cvt_bf16(const float* __restrict__ in,
                                                  unsigned short* __restrict__ out,
                                                  int n) {
    int i = (blockIdx.x * 256 + threadIdx.x) * 4;
    if (i + 3 < n) {
        float4 f = *(const float4*)(in + i);
        ushort4 o;
        o.x = f2bfu(f.x); o.y = f2bfu(f.y); o.z = f2bfu(f.z); o.w = f2bfu(f.w);
        *(ushort4*)(out + i) = o;
    }
}

// ---------------- 2) weight f32 [K,N] -> bf16 transposed [N,K] ----------------
__global__ __launch_bounds__(256) void k_cvt_w_t(const float* __restrict__ in,
                                                 unsigned short* __restrict__ out,
                                                 int K, int N) {
    int idx = blockIdx.x * 256 + threadIdx.x;
    if (idx < N * K) {
        int col = idx / K;
        int k   = idx % K;
        out[idx] = f2bfu(in[k * N + col]);
    }
}

// ---------------- 3) fused QKV projection GEMM ----------------
// Block = 4 waves, tile = 32 rows x 256 cols. X tile (32x32) staged to LDS
// (async double-buffered), shared by all 4 waves. Each wave: 32x64 via
// 8 f32 accumulators, B fragments double-buffered from global.
__global__ __launch_bounds__(128) void k_qkv(const unsigned short* __restrict__ Xbf,
                                             const unsigned short* __restrict__ WqT,
                                             const unsigned short* __restrict__ WkT,
                                             const unsigned short* __restrict__ WvT,
                                             unsigned short* __restrict__ Q,
                                             unsigned short* __restrict__ K,
                                             unsigned short* __restrict__ Vt) {
    const int mt   = blockIdx.x;            // 0..255  (32-row tile)
    const int nt   = blockIdx.y;            // 0..11   (256-col span across Wq|Wk|Wv)
    const int tid  = threadIdx.x;
    const int wid  = tid >> 5;
    const int lane = tid & 31;
    const int lh   = lane >> 4;
    const int ln   = lane & 15;

    const int wsel = nt >> 2;               // 0=Q 1=K 2=V (uniform per block)
    const unsigned short* W = (wsel == 0) ? WqT : (wsel == 1) ? WkT : WvT;
    const int ncol0 = (nt & 3) * 256 + wid * 64;   // col base within weight

    __shared__ __align__(16) unsigned short As[2][32 * 32];  // [buf][row*32 + k]

    // staging: 128 lanes x 16B = 2KB = one 32x32 bf16 tile
    const int srow = tid >> 2;              // 0..31
    const int soff = (tid & 3) * 8;         // element offset within 32-elem row
    const unsigned short* gsrc = Xbf + (mt * 32 + srow) * D_MODEL + soff;

    // prologue: stage k=0 into buffer 0
#if HAVE_ASYNC_LDS
    async_cp16(gsrc, &As[0][srow * 32 + soff]);
#endif

    // B fragment base pointers (per N-subtile)
    const unsigned short* wp[4];
#pragma unroll
    for (int j = 0; j < 4; ++j)
        wp[j] = W + (ncol0 + j * 16 + ln) * D_MODEL + lh * 8;

    Frag bc[4];
#pragma unroll
    for (int j = 0; j < 4; ++j) {
        bc[j].q[0] = *(const uint4*)(wp[j]);
        bc[j].q[1] = *(const uint4*)(wp[j] + 16);
    }

    v8f acc[8];
#pragma unroll
    for (int j = 0; j < 8; ++j) acc[j] = zero8();

    for (int k = 0; k < D_MODEL; k += 32) {
        const int buf = (k >> 5) & 1;
#if HAVE_ASYNC_LDS
        // stage next k-tile into the other buffer (always; trailing read stays
        // inside the workspace). Previous-iteration end barrier guarantees no
        // wave is still reading buf^1.
        async_cp16(gsrc + k + 32, &As[buf ^ 1][srow * 32 + soff]);
        asm volatile("s_wait_asynccnt 0x1" ::: "memory"); // my stage for buf done
        __syncthreads();                                  // everyone's stage done
#else
        __syncthreads();
        {   // sync fallback: global -> reg -> LDS (single buffer)
            uint4 t = *(const uint4*)(gsrc + k);
            *(uint4*)&As[buf][srow * 32 + soff] = t;
        }
        __syncthreads();
#endif
        // A fragments (two 16-row subtiles) from LDS
        Frag a0, a1;
        a0.q[0] = *(const uint4*)&As[buf][(ln)      * 32 +      lh * 8];
        a0.q[1] = *(const uint4*)&As[buf][(ln)      * 32 + 16 + lh * 8];
        a1.q[0] = *(const uint4*)&As[buf][(16 + ln) * 32 +      lh * 8];
        a1.q[1] = *(const uint4*)&As[buf][(16 + ln) * 32 + 16 + lh * 8];

        // prefetch next B fragments before issuing WMMAs on current ones
        Frag bn[4];
        if (k + 32 < D_MODEL) {
#pragma unroll
            for (int j = 0; j < 4; ++j) {
                bn[j].q[0] = *(const uint4*)(wp[j] + k + 32);
                bn[j].q[1] = *(const uint4*)(wp[j] + k + 48);
            }
        }

#pragma unroll
        for (int j = 0; j < 4; ++j) acc[j]     = wmma_bf16(a0, bc[j], acc[j]);
#pragma unroll
        for (int j = 0; j < 4; ++j) acc[4 + j] = wmma_bf16(a1, bc[j], acc[4 + j]);

        if (k + 32 < D_MODEL) {
#pragma unroll
            for (int j = 0; j < 4; ++j) bc[j] = bn[j];
        }
        __syncthreads();   // all waves done reading buf before it is restaged
    }

    // scatter C into Q/K/Vt layouts
#pragma unroll
    for (int ms = 0; ms < 2; ++ms) {
#pragma unroll
        for (int j = 0; j < 4; ++j) {
#pragma unroll
            for (int i = 0; i < 8; ++i) {
                const int r  = mt * 32 + ms * 16 + i + lh * 8;   // token row
                const int c  = ncol0 + j * 16 + ln;              // 0..1023
                const int bb = r >> 11;
                const int s  = r & 2047;
                const int h  = c >> 6;
                const int d  = c & 63;
                const unsigned short val = f2bfu(acc[ms * 4 + j][i]);
                if (wsel == 0)      Q [((bb * NUM_HEADS + h) * SEQ + s) * DK + d] = val;
                else if (wsel == 1) K [((bb * NUM_HEADS + h) * SEQ + s) * DK + d] = val;
                else                Vt[((bb * NUM_HEADS + h) * DK + d) * SEQ + s] = val;
            }
        }
    }
}

// ---------------- 4) flash attention (causal), one 16-query tile per wave ----------------
__global__ __launch_bounds__(32) void k_attn(const unsigned short* __restrict__ Q,
                                             const unsigned short* __restrict__ Kb,
                                             const unsigned short* __restrict__ Vt,
                                             unsigned short* __restrict__ Cc) {
    const int qt   = blockIdx.x;            // 0..127
    const int bh   = blockIdx.y;            // 0..63
    const int lane = threadIdx.x;
    const int lh   = lane >> 4;
    const int ln   = lane & 15;

    const unsigned short* Qh = Q  + (size_t)bh * SEQ * DK;
    const unsigned short* Kh = Kb + (size_t)bh * SEQ * DK;
    const unsigned short* Vh = Vt + (size_t)bh * DK * SEQ;

    __shared__ __align__(16) unsigned short P[32 * 16];   // col-major [col][row]

    const int qr   = qt * 16;
    const int qrow = qr + ln;
    Frag qa0, qa1;
    qa0.q[0] = *(const uint4*)(Qh + qrow * DK +  0 + lh * 16);
    qa0.q[1] = *(const uint4*)(Qh + qrow * DK +  8 + lh * 16);
    qa1.q[0] = *(const uint4*)(Qh + qrow * DK + 32 + lh * 16);
    qa1.q[1] = *(const uint4*)(Qh + qrow * DK + 40 + lh * 16);

    float m[8], l[8];
#pragma unroll
    for (int i = 0; i < 8; ++i) { m[i] = -1e30f; l[i] = 0.f; }
    v8f o[4];
#pragma unroll
    for (int j = 0; j < 4; ++j) o[j] = zero8();

    const float scale = 0.125f;
    const int kend = qr + 16;

    // K fragments for a 32-key block: [tile(2)][dk-half(2)]
    Frag kc[4];
    {
        const unsigned short* kp0 = Kh + (0 + ln) * DK + lh * 16;
        const unsigned short* kp1 = Kh + (16 + ln) * DK + lh * 16;
        kc[0].q[0] = *(const uint4*)(kp0);      kc[0].q[1] = *(const uint4*)(kp0 + 8);
        kc[1].q[0] = *(const uint4*)(kp0 + 32); kc[1].q[1] = *(const uint4*)(kp0 + 40);
        kc[2].q[0] = *(const uint4*)(kp1);      kc[2].q[1] = *(const uint4*)(kp1 + 8);
        kc[3].q[0] = *(const uint4*)(kp1 + 32); kc[3].q[1] = *(const uint4*)(kp1 + 40);
    }

    for (int k0 = 0; k0 < kend; k0 += 32) {
        // prefetch next key block's K fragments (uniform branch per wave)
        Frag kn[4];
        const bool more = (k0 + 32) < kend;
        if (more) {
            const unsigned short* kp0 = Kh + (k0 + 32 + ln) * DK + lh * 16;
            const unsigned short* kp1 = Kh + (k0 + 48 + ln) * DK + lh * 16;
            kn[0].q[0] = *(const uint4*)(kp0);      kn[0].q[1] = *(const uint4*)(kp0 + 8);
            kn[1].q[0] = *(const uint4*)(kp0 + 32); kn[1].q[1] = *(const uint4*)(kp0 + 40);
            kn[2].q[0] = *(const uint4*)(kp1);      kn[2].q[1] = *(const uint4*)(kp1 + 8);
            kn[3].q[0] = *(const uint4*)(kp1 + 32); kn[3].q[1] = *(const uint4*)(kp1 + 40);
        }

        // ---- scores ----
        v8f s0 = zero8(), s1 = zero8();
        s0 = wmma_bf16(qa0, kc[0], s0);
        s0 = wmma_bf16(qa1, kc[1], s0);
        s1 = wmma_bf16(qa0, kc[2], s1);
        s1 = wmma_bf16(qa1, kc[3], s1);

        // issue V fragment loads now; latency hides under softmax VALU work
        Frag vb[4];
#pragma unroll
        for (int j = 0; j < 4; ++j) {
            const unsigned short* vp = Vh + (j * 16 + ln) * SEQ + k0 + lh * 8;
            vb[j].q[0] = *(const uint4*)(vp);
            vb[j].q[1] = *(const uint4*)(vp + 16);
        }

        // ---- scale + causal mask + online softmax ----
#pragma unroll
        for (int i = 0; i < 8; ++i) {
            const int r  = qr + i + lh * 8;
            const int c0 = k0 + ln;
            const int c1 = k0 + 16 + ln;
            float a0 = (c0 <= r) ? s0[i] * scale : -1e30f;
            float a1 = (c1 <= r) ? s1[i] * scale : -1e30f;

            float rmax = fmaxf(a0, a1);
            rmax = fmaxf(rmax, __shfl_xor(rmax, 1, 32));
            rmax = fmaxf(rmax, __shfl_xor(rmax, 2, 32));
            rmax = fmaxf(rmax, __shfl_xor(rmax, 4, 32));
            rmax = fmaxf(rmax, __shfl_xor(rmax, 8, 32));

            const float mn    = fmaxf(m[i], rmax);
            const float alpha = __expf(m[i] - mn);
            m[i] = mn;

            const float p0 = __expf(a0 - mn);
            const float p1 = __expf(a1 - mn);
            s0[i] = p0; s1[i] = p1;

            float rs = p0 + p1;
            rs += __shfl_xor(rs, 1, 32);
            rs += __shfl_xor(rs, 2, 32);
            rs += __shfl_xor(rs, 4, 32);
            rs += __shfl_xor(rs, 8, 32);
            l[i] = l[i] * alpha + rs;

#pragma unroll
            for (int j = 0; j < 4; ++j) o[j][i] *= alpha;
        }

        // ---- P: C layout -> A layout via LDS ----
        {
            uint4 pk0, pk1;
            unsigned int* w0 = (unsigned int*)&pk0;
            unsigned int* w1 = (unsigned int*)&pk1;
#pragma unroll
            for (int ii = 0; ii < 4; ++ii) {
                w0[ii] = (unsigned int)f2bfu(s0[2 * ii]) | ((unsigned int)f2bfu(s0[2 * ii + 1]) << 16);
                w1[ii] = (unsigned int)f2bfu(s1[2 * ii]) | ((unsigned int)f2bfu(s1[2 * ii + 1]) << 16);
            }
            *(uint4*)&P[(ln)      * 16 + lh * 8] = pk0;
            *(uint4*)&P[(16 + ln) * 16 + lh * 8] = pk1;
        }
        asm volatile("s_wait_dscnt 0" ::: "memory");

        Frag pf;
#pragma unroll
        for (int c = 0; c < 8; ++c) {
            pf.us[c]     = P[(lh * 8 + c)      * 16 + ln];
            pf.us[8 + c] = P[(16 + lh * 8 + c) * 16 + ln];
        }

        // ---- O += P @ V ----
#pragma unroll
        for (int j = 0; j < 4; ++j) o[j] = wmma_bf16(pf, vb[j], o[j]);

        if (more) {
#pragma unroll
            for (int j = 0; j < 4; ++j) kc[j] = kn[j];
        }
    }

    // ---- normalize, store concat [B, S, H*64] bf16 ----
    const int b = bh >> 4, h = bh & 15;
#pragma unroll
    for (int j = 0; j < 4; ++j) {
#pragma unroll
        for (int i = 0; i < 8; ++i) {
            const int s   = qr + i + lh * 8;
            const int col = h * DK + j * 16 + ln;
            const float v = o[j][i] / l[i];
            Cc[((size_t)b * SEQ + s) * (NUM_HEADS * DK) + col] = f2bfu(v);
        }
    }
}

// ---------------- 5) output projection GEMM + bias (f32 out) ----------------
// One wave per 32x64 tile, fragments double-buffered from global.
__global__ __launch_bounds__(32) void k_out(const unsigned short* __restrict__ Cc,
                                            const unsigned short* __restrict__ WoT,
                                            const float* __restrict__ bO,
                                            float* __restrict__ Out) {
    const int mt   = blockIdx.x;            // 0..255
    const int nt   = blockIdx.y;            // 0..15
    const int lane = threadIdx.x;
    const int lh   = lane >> 4;
    const int ln   = lane & 15;
    const int ncol0 = nt * 64;

    v8f acc[8];
#pragma unroll
    for (int j = 0; j < 8; ++j) acc[j] = zero8();

    const unsigned short* ar0 = Cc + (mt * 32 + ln)      * D_MODEL;
    const unsigned short* ar1 = Cc + (mt * 32 + 16 + ln) * D_MODEL;
    const unsigned short* wp[4];
#pragma unroll
    for (int j = 0; j < 4; ++j)
        wp[j] = WoT + (ncol0 + j * 16 + ln) * D_MODEL + lh * 8;

    Frag a0c, a1c, bc[4];
    a0c.q[0] = *(const uint4*)(ar0 +      lh * 8);
    a0c.q[1] = *(const uint4*)(ar0 + 16 + lh * 8);
    a1c.q[0] = *(const uint4*)(ar1 +      lh * 8);
    a1c.q[1] = *(const uint4*)(ar1 + 16 + lh * 8);
#pragma unroll
    for (int j = 0; j < 4; ++j) {
        bc[j].q[0] = *(const uint4*)(wp[j]);
        bc[j].q[1] = *(const uint4*)(wp[j] + 16);
    }

    for (int k = 0; k < D_MODEL; k += 32) {
        Frag a0n, a1n, bn[4];
        const bool more = (k + 32) < D_MODEL;
        if (more) {
            a0n.q[0] = *(const uint4*)(ar0 + k + 32 +      lh * 8);
            a0n.q[1] = *(const uint4*)(ar0 + k + 48 + lh * 8);
            a1n.q[0] = *(const uint4*)(ar1 + k + 32 +      lh * 8);
            a1n.q[1] = *(const uint4*)(ar1 + k + 48 + lh * 8);
#pragma unroll
            for (int j = 0; j < 4; ++j) {
                bn[j].q[0] = *(const uint4*)(wp[j] + k + 32);
                bn[j].q[1] = *(const uint4*)(wp[j] + k + 48);
            }
            __builtin_prefetch(ar0 + k + 96, 0, 1);
        }

#pragma unroll
        for (int j = 0; j < 4; ++j) acc[j]     = wmma_bf16(a0c, bc[j], acc[j]);
#pragma unroll
        for (int j = 0; j < 4; ++j) acc[4 + j] = wmma_bf16(a1c, bc[j], acc[4 + j]);

        if (more) {
            a0c = a0n; a1c = a1n;
#pragma unroll
            for (int j = 0; j < 4; ++j) bc[j] = bn[j];
        }
    }

#pragma unroll
    for (int ms = 0; ms < 2; ++ms) {
#pragma unroll
        for (int j = 0; j < 4; ++j) {
#pragma unroll
            for (int i = 0; i < 8; ++i) {
                const int r = mt * 32 + ms * 16 + i + lh * 8;
                const int c = ncol0 + j * 16 + ln;
                Out[(size_t)r * D_MODEL + c] = acc[ms * 4 + j][i] + bO[c];
            }
        }
    }
}

// ---------------- launch ----------------
extern "C" void kernel_launch(void* const* d_in, const int* in_sizes, int n_in,
                              void* d_out, int out_size, void* d_ws, size_t ws_size,
                              hipStream_t stream) {
    const float* XKV = (const float*)d_in[0];
    const float* WQ  = (const float*)d_in[1];
    const float* WK  = (const float*)d_in[2];
    const float* WV  = (const float*)d_in[3];
    const float* WO  = (const float*)d_in[4];
    const float* bO  = (const float*)d_in[5];
    float* Out = (float*)d_out;

    char* ws = (char*)d_ws;                     // needs ~92 MB
    const size_t XBF_B = (size_t)ROWS * D_MODEL * 2;
    const size_t W_B   = (size_t)D_MODEL * D_MODEL * 2;
    const size_t QKV_B = (size_t)BATCH * NUM_HEADS * SEQ * DK * 2;

    unsigned short* Xbf = (unsigned short*)(ws);
    unsigned short* WqT = (unsigned short*)(ws + XBF_B);
    unsigned short* WkT = (unsigned short*)(ws + XBF_B + W_B);
    unsigned short* WvT = (unsigned short*)(ws + XBF_B + 2 * W_B);
    unsigned short* WoT = (unsigned short*)(ws + XBF_B + 3 * W_B);
    unsigned short* Qb  = (unsigned short*)(ws + XBF_B + 4 * W_B);
    unsigned short* Kb  = (unsigned short*)(ws + XBF_B + 4 * W_B + QKV_B);
    unsigned short* Vt  = (unsigned short*)(ws + XBF_B + 4 * W_B + 2 * QKV_B);
    unsigned short* Cc  = (unsigned short*)(ws + XBF_B + 4 * W_B + 3 * QKV_B);

    k_cvt_bf16<<<(ROWS * D_MODEL) / (256 * 4), 256, 0, stream>>>(XKV, Xbf, ROWS * D_MODEL);
    const int WN = (D_MODEL * D_MODEL + 255) / 256;
    k_cvt_w_t<<<WN, 256, 0, stream>>>(WQ, WqT, D_MODEL, D_MODEL);
    k_cvt_w_t<<<WN, 256, 0, stream>>>(WK, WkT, D_MODEL, D_MODEL);
    k_cvt_w_t<<<WN, 256, 0, stream>>>(WV, WvT, D_MODEL, D_MODEL);
    k_qkv<<<dim3(ROWS / 32, 12), 128, 0, stream>>>(Xbf, WqT, WkT, WvT, Qb, Kb, Vt);
    k_attn<<<dim3(SEQ / 16, BATCH * NUM_HEADS), 32, 0, stream>>>(Qb, Kb, Vt, Cc);
    k_out<<<dim3(ROWS / 32, D_MODEL / 64), 32, 0, stream>>>(Cc, WoT, bO, Out);
    (void)in_sizes; (void)n_in; (void)out_size; (void)ws_size;
}
